// BiCameralCRSN_24902220382645
// MI455X (gfx1250) — compile-verified, matrix operand order checked by reading, not run on hardware
//
#include <hip/hip_runtime.h>
#include <hip/hip_bf16.h>
#include <math.h>

// ---------------- problem constants ----------------
constexpr int B = 2048, D = 512, V = 32000, S = 32, DEPTH = 10;
constexpr int TWO_D = 2 * D; // 1024

typedef __attribute__((ext_vector_type(16))) _Float16 v16h;
typedef __attribute__((ext_vector_type(8)))  float    v8f;

// --- CDNA5 async memory->LDS staging (ASYNCcnt-tracked, no VGPR data) ---
__device__ __forceinline__ void async_copy_b128(uint32_t lds_byte_addr,
                                                const void* gaddr) {
  asm volatile("global_load_async_to_lds_b128 %0, %1, off"
               :: "v"(lds_byte_addr), "v"((uint64_t)(uintptr_t)gaddr)
               : "memory");
}
__device__ __forceinline__ void wait_asynccnt0() {
  asm volatile("s_wait_asynccnt 0x0" ::: "memory");
}

// =====================================================================
// K0: initialize all persistent state (outputs + ws scalars)
// grid covers stack (largest region) exactly: B*S*2D threads
// =====================================================================
__global__ void init_state(float* __restrict__ zo_r, float* __restrict__ zo_i,
                           float* __restrict__ stack, float* __restrict__ ptrS,
                           float* __restrict__ halt, float* __restrict__ remain,
                           float* __restrict__ pond_loss /* 2 floats */) {
  size_t i = (size_t)blockIdx.x * 256 + threadIdx.x;
  stack[i] = 0.f;
  if (i < (size_t)B * D) { zo_r[i] = 0.f; zo_i[i] = 0.f; }
  if (i < (size_t)B * S) ptrS[i] = ((i & (S - 1)) == 0) ? 1.f : 0.f;
  if (i < (size_t)B)     { halt[i] = 0.f; remain[i] = 1.f; }
  if (i < 2)             pond_loss[i] = 0.f;
}

// =====================================================================
// K1: pack complex weights into one f16 block matrix, stored [n][k]
// =====================================================================
__global__ void pack_weights(const float* __restrict__ Wr, const float* __restrict__ Wi,
                             _Float16* __restrict__ Wc) {
  int i = blockIdx.x * 256 + threadIdx.x;       // over 1024*1024
  int n = i >> 10, k = i & (TWO_D - 1);
  float v;
  if (n < D) v = (k < D) ? Wr[n * D + k] : -Wi[n * D + (k - D)];
  else       v = (k < D) ? Wi[(n - D) * D + k] : Wr[(n - D) * D + (k - D)];
  Wc[i] = (_Float16)v;
}

// =====================================================================
// K2: embedding gather + polar->cartesian, produce f16 z_cat [B][2D]
// =====================================================================
__global__ void embed_init(const int* __restrict__ x_idx,
                           const float* __restrict__ emb_mag,
                           const float* __restrict__ emb_phase,
                           _Float16* __restrict__ z) {
  int i = blockIdx.x * 256 + threadIdx.x;       // B*2D
  int b = i >> 10, j = i & (TWO_D - 1);
  int c = j & (D - 1);
  int v = x_idx[b];
  float r = emb_mag[(size_t)v * D + c];
  float t = emb_phase[(size_t)v * D + c];
  float val = (j < D) ? r * cosf(t) : r * sinf(t);
  z[(size_t)b * TWO_D + j] = (_Float16)val;
}

// =====================================================================
// K3: recurrent GEMM  p[B][1024] = z[B][1024](f16) @ Wc (f16, [n][k])
// 256 threads = 8 waves; M-tile=16 async-staged into LDS; wave -> 128 N
// =====================================================================
__global__ void gemm_recurrent(const _Float16* __restrict__ z,
                               const _Float16* __restrict__ Wc,
                               float* __restrict__ p) {
  __shared__ __align__(32) _Float16 Atile[16 * TWO_D]; // 32 KB
  const int mbase = blockIdx.x * 16;
  const int tid = threadIdx.x;

  {
    const uint32_t lds_base = (uint32_t)(uintptr_t)Atile;
    const _Float16* gsrc = z + (size_t)mbase * TWO_D;
    for (int i = tid; i < 16 * TWO_D / 8; i += 256)
      async_copy_b128(lds_base + i * 16, gsrc + i * 8);
    wait_asynccnt0();
  }
  __syncthreads();

  const int wave = tid >> 5, lane = tid & 31;
  const int m = lane & 15;
  const int koff = (lane >> 4) * 16;
  const int nbase = wave * 128;

  v8f acc[8];
#pragma unroll
  for (int t = 0; t < 8; ++t) acc[t] = (v8f){0.f,0.f,0.f,0.f,0.f,0.f,0.f,0.f};

  for (int kb = 0; kb < TWO_D; kb += 32) {
    v16h a = *(const v16h*)(Atile + m * TWO_D + kb + koff);
#pragma unroll
    for (int t = 0; t < 8; ++t) {
      int n = nbase + t * 16 + (lane & 15);
      v16h bf = *(const v16h*)(Wc + (size_t)n * TWO_D + kb + koff);
      acc[t] = __builtin_amdgcn_wmma_f32_16x16x32_f16(
          false, a, false, bf, (short)0, acc[t], false, false);
    }
  }
  const int mrow = (lane >> 4) * 8;
#pragma unroll
  for (int t = 0; t < 8; ++t) {
    int n = nbase + t * 16 + (lane & 15);
#pragma unroll
    for (int j = 0; j < 8; ++j)
      p[(size_t)(mbase + mrow + j) * TWO_D + n] = acc[t][j];
  }
}

// =====================================================================
// K4: fused per-row step: mag-LN, phase renorm, relu-modulus, ctrl
// softmax, ptr roll, stack RMW + weighted read (single streaming pass),
// ACT halting, zo accumulation, next-z (f16). One block per batch row.
// =====================================================================
__global__ void step_update(const float* __restrict__ p,
                            const float* __restrict__ ln_scale, const float* __restrict__ ln_shift,
                            const float* __restrict__ mr_bias,
                            const float* __restrict__ W_halt, const float* __restrict__ b_halt,
                            const float* __restrict__ W_ctrl, const float* __restrict__ b_ctrl,
                            float* __restrict__ stack, float* __restrict__ ptrS,
                            float* __restrict__ zo_r, float* __restrict__ zo_i,
                            float* __restrict__ halt_prob, float* __restrict__ remain,
                            float* __restrict__ ponder,
                            _Float16* __restrict__ z_next, float* __restrict__ fsr,
                            int step) {
  __shared__ float zsh[TWO_D];
  __shared__ float red[256], red2[256];
  __shared__ float4 red4[256];
  __shared__ float sptr[S], snew[S], swm[S];
  __shared__ float sc[6]; // c0 c1 c2 h invsum -
  const int b = blockIdx.x, tid = threadIdx.x;
  const float* prow = p + (size_t)b * TWO_D;

  // --- magnitude + stats (2 complex dims / thread) ---
  float prv[2], piv[2], mg[2];
  float lsum = 0.f, lsq = 0.f;
#pragma unroll
  for (int u = 0; u < 2; ++u) {
    int c = tid * 2 + u;
    prv[u] = prow[c]; piv[u] = prow[D + c];
    mg[u] = sqrtf(prv[u] * prv[u] + piv[u] * piv[u]) + 1e-6f;
    lsum += mg[u]; lsq += mg[u] * mg[u];
  }
  red[tid] = lsum; red2[tid] = lsq; __syncthreads();
  for (int s = 128; s > 0; s >>= 1) {
    if (tid < s) { red[tid] += red[tid + s]; red2[tid] += red2[tid + s]; }
    __syncthreads();
  }
  const float mean = red[0] * (1.f / D);
  const float var = (red2[0] - (float)D * mean * mean) * (1.f / (D - 1));
  const float inv = rsqrtf(var + 1e-6f);

  // --- layernorm on magnitude, rotate back, relu modulus ---
#pragma unroll
  for (int u = 0; u < 2; ++u) {
    int c = tid * 2 + u;
    float nm = (mg[u] - mean) * inv * ln_scale[c] + ln_shift[c];
    float ih = 1.f / mg[u];
    float p2r = nm * prv[u] * ih, p2i = nm * piv[u] * ih;
    float n2 = sqrtf(p2r * p2r + p2i * p2i) + 1e-6f;
    float scl = fmaxf(n2 + mr_bias[c], 0.f) / n2;
    zsh[c] = p2r * scl;
    zsh[D + c] = p2i * scl;
  }
  __syncthreads();

  // --- 4 fused dot products: ctrl0..2, halt ---
  float4 part = {0.f, 0.f, 0.f, 0.f};
#pragma unroll
  for (int u = 0; u < 4; ++u) {
    int j = tid * 4 + u;
    float zv = zsh[j];
    part.x += zv * W_ctrl[j];
    part.y += zv * W_ctrl[TWO_D + j];
    part.z += zv * W_ctrl[2 * TWO_D + j];
    part.w += zv * W_halt[j];
  }
  red4[tid] = part; __syncthreads();
  for (int s = 128; s > 0; s >>= 1) {
    if (tid < s) {
      float4 a = red4[tid], c = red4[tid + s];
      a.x += c.x; a.y += c.y; a.z += c.z; a.w += c.w;
      red4[tid] = a;
    }
    __syncthreads();
  }
  if (tid == 0) {
    float l0 = red4[0].x + b_ctrl[0], l1 = red4[0].y + b_ctrl[1], l2 = red4[0].z + b_ctrl[2];
    float mx = fmaxf(l0, fmaxf(l1, l2));
    float e0 = expf(l0 - mx), e1 = expf(l1 - mx), e2 = expf(l2 - mx);
    float is = 1.f / (e0 + e1 + e2);
    sc[0] = e0 * is; sc[1] = e1 * is; sc[2] = e2 * is;
    sc[3] = 1.f / (1.f + expf(-(red4[0].w + b_halt[0])));
  }
  if (tid < S) sptr[tid] = ptrS[(size_t)b * S + tid];
  __syncthreads();
  if (tid < S) {
    float up = sptr[(tid + S - 1) & (S - 1)];
    float dn = sptr[(tid + 1) & (S - 1)];
    snew[tid] = sc[0] * up + sc[1] * dn + sc[2] * sptr[tid];
    swm[tid]  = sc[0] * up;
  }
  __syncthreads();
  if (tid == 0) {
    float s = 0.f;
    for (int i = 0; i < S; ++i) s += snew[i];
    sc[4] = 1.f / (s + 1e-6f);
  }
  __syncthreads();
  const float invsum = sc[4];

  // --- fused stack RMW + weighted read (one load + one store / elem) ---
  const int j4 = tid * 4;
  float4 z4v = *(float4*)(zsh + j4);
  const float* z4 = (const float*)&z4v;
  float rd[4] = {0.f, 0.f, 0.f, 0.f};
  float* srow = stack + (size_t)b * S * TWO_D + j4;
#pragma unroll 4
  for (int s = 0; s < S; ++s) {
    float wm = swm[s];
    float npv = snew[s] * invsum;
    float4 stv = *(float4*)(srow + (size_t)s * TWO_D);
    float* st = (float*)&stv;
#pragma unroll
    for (int u = 0; u < 4; ++u) {
      st[u] = st[u] * (1.f - wm) + z4[u] * wm;
      rd[u] += st[u] * npv;
    }
    *(float4*)(srow + (size_t)s * TWO_D) = stv;
  }

  // --- ACT halting + zo accumulation + state for next step ---
  const float hp = halt_prob[b], rem = remain[b];
  const float still = (hp < 0.99f) ? 1.f : 0.f;
  float pp = fminf(sc[3] * still, rem);
  if (step == DEPTH - 1) pp = rem;
#pragma unroll
  for (int u = 0; u < 4; ++u) {
    int j = j4 + u;
    float zc = zsh[j] + rd[u];
    fsr[(size_t)b * TWO_D + j] = rd[u];
    if (j < D) zo_r[(size_t)b * D + j] += pp * zc;
    else       zo_i[(size_t)b * D + (j - D)] += pp * zc;
    z_next[(size_t)b * TWO_D + j] = (_Float16)zc;
  }
  if (tid == 0) {
    halt_prob[b] = hp + pp;
    remain[b] = rem - pp;
    atomicAdd(ponder, still * (1.f / (float)B));
  }
  if (tid < S) ptrS[(size_t)b * S + tid] = snew[tid] * invsum;
}

// =====================================================================
// K5: VQ argmin (straight-through forward == codebook row), losses,
// final feature assembly (f16 for the decoder WMMA)
// =====================================================================
__global__ void vq_feats(const float* __restrict__ zo_r, const float* __restrict__ zo_i,
                         const float* __restrict__ fsr,
                         const float* __restrict__ cb_syn, const float* __restrict__ cb_sem,
                         float* __restrict__ idx_syn, float* __restrict__ idx_sem,
                         float* __restrict__ loss, _Float16* __restrict__ feats) {
  __shared__ float za[TWO_D], zb[TWO_D];
  __shared__ float red[256];
  __shared__ int sidx[2];
  const int b = blockIdx.x, tid = threadIdx.x;
#pragma unroll
  for (int u = 0; u < 4; ++u) {
    int j = tid * 4 + u;
    za[j] = (j < D) ? zo_r[(size_t)b * D + j] : zo_i[(size_t)b * D + (j - D)];
    zb[j] = fsr[(size_t)b * TWO_D + j];
  }
  __syncthreads();

  float best = 3.4e38f; int bi = 0;
  for (int c = 0; c < 64; ++c) {
    float part = 0.f;
#pragma unroll
    for (int u = 0; u < 4; ++u) {
      int j = tid * 4 + u;
      float d = cb_syn[(size_t)c * TWO_D + j] - za[j];
      part += d * d;
    }
    red[tid] = part; __syncthreads();
    for (int s = 128; s > 0; s >>= 1) { if (tid < s) red[tid] += red[tid + s]; __syncthreads(); }
    if (tid == 0 && red[0] < best) { best = red[0]; bi = c; }
    __syncthreads();
  }
  if (tid == 0) {
    idx_syn[b] = (float)bi; sidx[0] = bi;
    atomicAdd(loss, 1.25f * best / ((float)B * TWO_D));
  }

  best = 3.4e38f; bi = 0;
  for (int c = 0; c < 128; ++c) {
    float part = 0.f;
#pragma unroll
    for (int u = 0; u < 4; ++u) {
      int j = tid * 4 + u;
      float d = cb_sem[(size_t)c * TWO_D + j] - zb[j];
      part += d * d;
    }
    red[tid] = part; __syncthreads();
    for (int s = 128; s > 0; s >>= 1) { if (tid < s) red[tid] += red[tid + s]; __syncthreads(); }
    if (tid == 0 && red[0] < best) { best = red[0]; bi = c; }
    __syncthreads();
  }
  if (tid == 0) {
    idx_sem[b] = (float)bi; sidx[1] = bi;
    atomicAdd(loss, 1.25f * best / ((float)B * TWO_D));
  }
  __syncthreads();
  const int cs = sidx[0], cm = sidx[1];
#pragma unroll
  for (int u = 0; u < 4; ++u) {
    int j = tid * 4 + u;
    float f = za[j] + cb_syn[(size_t)cs * TWO_D + j] + cb_sem[(size_t)cm * TWO_D + j];
    feats[(size_t)b * TWO_D + j] = (_Float16)f;
  }
}

// =====================================================================
// K6: decoder GEMM  logits[B][V] = feats(f16) @ W_dec.T + b_dec
// M-tile=16 async-staged into LDS; wave owns 2 x 16 columns (WMMA ILP);
// W_dec converted fp32->f16 on the fly (L2-resident), prefetch ahead.
// =====================================================================
__global__ void decoder_gemm(const _Float16* __restrict__ feats,
                             const float* __restrict__ Wd,
                             const float* __restrict__ bdec,
                             float* __restrict__ logits) {
  __shared__ __align__(32) _Float16 Atile[16 * TWO_D]; // 32 KB
  const int mbase = blockIdx.x * 16;
  const int tid = threadIdx.x;
  {
    const uint32_t lds_base = (uint32_t)(uintptr_t)Atile;
    const _Float16* gsrc = feats + (size_t)mbase * TWO_D;
    for (int i = tid; i < 16 * TWO_D / 8; i += 256)
      async_copy_b128(lds_base + i * 16, gsrc + i * 8);
    wait_asynccnt0();
  }
  __syncthreads();

  const int wave = tid >> 5, lane = tid & 31;
  const int m = lane & 15;
  const int koff = (lane >> 4) * 16;
  const int n0 = blockIdx.y * 256 + wave * 32 + (lane & 15);
  const int n1 = n0 + 16;

  v8f acc0 = {}, acc1 = {};
  const float* w0 = Wd + (size_t)n0 * TWO_D + koff;
  const float* w1 = Wd + (size_t)n1 * TWO_D + koff;

  for (int kb = 0; kb < TWO_D; kb += 32) {
    v16h a = *(const v16h*)(Atile + m * TWO_D + kb + koff);
    __builtin_prefetch(w0 + kb + 256, 0, 1);
    __builtin_prefetch(w1 + kb + 256, 0, 1);
    v16h b0, b1;
#pragma unroll
    for (int t = 0; t < 16; ++t) {
      b0[t] = (_Float16)w0[kb + t];
      b1[t] = (_Float16)w1[kb + t];
    }
    acc0 = __builtin_amdgcn_wmma_f32_16x16x32_f16(false, a, false, b0, (short)0, acc0, false, false);
    acc1 = __builtin_amdgcn_wmma_f32_16x16x32_f16(false, a, false, b1, (short)0, acc1, false, false);
  }
  const int mrow = (lane >> 4) * 8;
  const float bi0 = bdec[n0], bi1 = bdec[n1];
#pragma unroll
  for (int j = 0; j < 8; ++j) {
    logits[(size_t)(mbase + mrow + j) * V + n0] = acc0[j] + bi0;
    logits[(size_t)(mbase + mrow + j) * V + n1] = acc1[j] + bi1;
  }
}

// =====================================================================
// host launcher
// =====================================================================
extern "C" void kernel_launch(void* const* d_in, const int* in_sizes, int n_in,
                              void* d_out, int out_size, void* d_ws, size_t ws_size,
                              hipStream_t stream) {
  (void)in_sizes; (void)n_in; (void)out_size; (void)ws_size;

  const int*   x_idx     = (const int*)d_in[0];
  const float* emb_mag   = (const float*)d_in[1];
  const float* emb_phase = (const float*)d_in[2];
  const float* Wr        = (const float*)d_in[3];
  const float* Wi        = (const float*)d_in[4];
  const float* ln_scale  = (const float*)d_in[5];
  const float* ln_shift  = (const float*)d_in[6];
  const float* mr_bias   = (const float*)d_in[7];
  const float* W_halt    = (const float*)d_in[8];
  const float* b_halt    = (const float*)d_in[9];
  const float* W_ctrl    = (const float*)d_in[10];
  const float* b_ctrl    = (const float*)d_in[11];
  const float* cb_syn    = (const float*)d_in[12];
  const float* cb_sem    = (const float*)d_in[13];
  const float* W_dec     = (const float*)d_in[14];
  const float* b_dec     = (const float*)d_in[15];
  float* out = (float*)d_out;

  // output layout (flat float32, reference return order)
  const size_t o_logits = 0;
  const size_t o_zor   = o_logits + (size_t)B * V;
  const size_t o_zoi   = o_zor + (size_t)B * D;
  const size_t o_stack = o_zoi + (size_t)B * D;
  const size_t o_ptr   = o_stack + (size_t)B * S * TWO_D;
  const size_t o_isyn  = o_ptr + (size_t)B * S;
  const size_t o_isem  = o_isyn + (size_t)B;
  const size_t o_pond  = o_isem + (size_t)B;

  // workspace layout
  char* ws = (char*)d_ws;
  _Float16* Wc   = (_Float16*)ws; ws += (size_t)TWO_D * TWO_D * sizeof(_Float16);
  _Float16* zcat = (_Float16*)ws; ws += (size_t)B * TWO_D * sizeof(_Float16);
  float* pbuf    = (float*)ws;    ws += (size_t)B * TWO_D * sizeof(float);
  float* fsr     = (float*)ws;    ws += (size_t)B * TWO_D * sizeof(float);
  _Float16* feats= (_Float16*)ws; ws += (size_t)B * TWO_D * sizeof(_Float16);
  float* halt    = (float*)ws;    ws += (size_t)B * sizeof(float);
  float* remain  = (float*)ws;

  init_state<<<(B * S * TWO_D) / 256, 256, 0, stream>>>(
      out + o_zor, out + o_zoi, out + o_stack, out + o_ptr, halt, remain, out + o_pond);
  pack_weights<<<(TWO_D * TWO_D) / 256, 256, 0, stream>>>(Wr, Wi, Wc);
  embed_init<<<(B * TWO_D) / 256, 256, 0, stream>>>(x_idx, emb_mag, emb_phase, zcat);

  for (int step = 0; step < DEPTH; ++step) {
    gemm_recurrent<<<B / 16, 256, 0, stream>>>(zcat, Wc, pbuf);
    step_update<<<B, 256, 0, stream>>>(pbuf, ln_scale, ln_shift, mr_bias,
                                       W_halt, b_halt, W_ctrl, b_ctrl,
                                       out + o_stack, out + o_ptr,
                                       out + o_zor, out + o_zoi,
                                       halt, remain, out + o_pond,
                                       zcat, fsr, step);
  }

  vq_feats<<<B, 256, 0, stream>>>(out + o_zor, out + o_zoi, fsr, cb_syn, cb_sem,
                                  out + o_isyn, out + o_isem, out + o_pond + 1, feats);
  decoder_gemm<<<dim3(B / 16, V / 256), 256, 0, stream>>>(feats, W_dec, b_dec, out + o_logits);
}
